// Bottleneck_8057358647495
// MI455X (gfx1250) — compile-verified, matrix operand order checked by reading, not run on hardware
//
#include <hip/hip_runtime.h>
#include <math.h>

typedef __attribute__((ext_vector_type(16))) _Float16 v16h;
typedef __attribute__((ext_vector_type(8)))  float    v8f;

#define Bb   32
#define CIN0 256
#define Cc   64
#define Hh   56
#define Ww   56
#define HW   3136
#define NTt  196     // HW/16
#define CEM  32
#define CK   72
#define COUT 256

// ---------------------------------------------------------------------------
// Prep: swizzle w1 (64x256), c1_w (64x64), w3 (256x64) into WMMA A-fragment
// order: frag[((mt*KC+kc)*32+lane)*16+j] ; A layout: M = lane&15, hi=lane>>4,
// K = kc*32 + 16*(j>>3) + 8*hi + 2*((j>>1)&3) + (j&1)
// ---------------------------------------------------------------------------
__global__ void prep_frags(const float* __restrict__ w1,
                           const float* __restrict__ c1w,
                           const float* __restrict__ w3,
                           _Float16* __restrict__ w1f,
                           _Float16* __restrict__ c1f,
                           _Float16* __restrict__ w3f) {
  int tid = blockIdx.x * 256 + threadIdx.x;
  if (tid >= 36864) return;
  const float* src; _Float16* dst; int mt, kc, lane, j, cin, t;
  if (tid < 16384) {                       // w1: 4 mt, 8 kc
    t = tid; src = w1; dst = w1f; cin = 256;
    j = t & 15; lane = (t >> 4) & 31; kc = (t >> 9) & 7; mt = t >> 12;
  } else if (tid < 20480) {                // c1: 4 mt, 2 kc
    t = tid - 16384; src = c1w; dst = c1f; cin = 64;
    j = t & 15; lane = (t >> 4) & 31; kc = (t >> 9) & 1; mt = t >> 10;
  } else {                                 // w3: 16 mt, 2 kc
    t = tid - 20480; src = w3; dst = w3f; cin = 64;
    j = t & 15; lane = (t >> 4) & 31; kc = (t >> 9) & 1; mt = t >> 10;
  }
  int m  = lane & 15, hi = lane >> 4;
  int K  = kc * 32 + ((j >> 3) << 4) + (hi << 3) + (((j >> 1) & 3) << 1) + (j & 1);
  int mg = mt * 16 + m;
  dst[t] = (_Float16)src[mg * cin + K];
}

// ---------------------------------------------------------------------------
// WMMA GEMM: out[b][m][n] = affine( W(64 x CINp) @ src[b](CINp x HW) )
// block = 128 threads (4 waves), each wave owns one 16-row m-tile; 16-col tile
// ---------------------------------------------------------------------------
template <int CINp>
__global__ void __launch_bounds__(128)
gemm64(const float* __restrict__ src, const _Float16* __restrict__ wf,
       const float* __restrict__ ga, const float* __restrict__ bb,
       float* __restrict__ out) {
  constexpr int KC = CINp / 32;
  __shared__ _Float16 Xt[16][CINp];               // transposed activation tile
  int b = blockIdx.x / NTt, nt = blockIdx.x % NTt;
  int n0 = nt * 16;
  const float* xb = src + (size_t)b * CINp * HW;
  int nn = threadIdx.x & 15;
  for (int c0 = threadIdx.x >> 4; c0 < CINp; c0 += 8)
    Xt[nn][c0] = (_Float16)xb[c0 * HW + n0 + nn];
  __syncthreads();

  int wave = threadIdx.x >> 5, lane = threadIdx.x & 31;
  int hi = lane >> 4, n = lane & 15, mt = wave;
  v8f acc = {};
  for (int kc = 0; kc < KC; ++kc) {
    v16h A = *(const v16h*)(wf + (((mt * KC + kc) * 32 + lane) << 4));
    v16h Bf;
    const _Float16* xr = &Xt[n][kc * 32 + (hi << 3)];
#pragma unroll
    for (int p = 0; p < 8; ++p) {
      int ko = ((p >> 2) << 4) + ((p & 3) << 1);
      Bf[2 * p]     = xr[ko];
      Bf[2 * p + 1] = xr[ko + 1];
    }
    acc = __builtin_amdgcn_wmma_f32_16x16x32_f16(false, A, false, Bf,
                                                 (short)0, acc, false, false);
  }
  float* ob = out + (size_t)b * Cc * HW;
#pragma unroll
  for (int i = 0; i < 8; ++i) {
    int m = mt * 16 + (hi << 3) + i;              // C/D: VGPR i -> M=i(+8 for hi)
    ob[m * HW + n0 + n] = acc[i] * ga[m] + bb[m];
  }
}

// ---------------------------------------------------------------------------
// Expand GEMM 64->256 + BN + relu + residual. 16 m-tiles: 4 waves x 4 iters.
// ---------------------------------------------------------------------------
__global__ void __launch_bounds__(128)
k_expand(const float* __restrict__ cot, const _Float16* __restrict__ w3f,
         const float* __restrict__ g3, const float* __restrict__ b3,
         const float* __restrict__ xin, float* __restrict__ out) {
  __shared__ _Float16 Xt[16][64];
  int b = blockIdx.x / NTt, nt = blockIdx.x % NTt;
  int n0 = nt * 16;
  const float* cb = cot + (size_t)b * Cc * HW;
  int nn = threadIdx.x & 15;
  for (int c0 = threadIdx.x >> 4; c0 < 64; c0 += 8)
    Xt[nn][c0] = (_Float16)cb[c0 * HW + n0 + nn];
  __syncthreads();

  int wave = threadIdx.x >> 5, lane = threadIdx.x & 31;
  int hi = lane >> 4, n = lane & 15;
  v16h Bfr[2];
#pragma unroll
  for (int kc = 0; kc < 2; ++kc) {
    const _Float16* xr = &Xt[n][kc * 32 + (hi << 3)];
#pragma unroll
    for (int p = 0; p < 8; ++p) {
      int ko = ((p >> 2) << 4) + ((p & 3) << 1);
      Bfr[kc][2 * p]     = xr[ko];
      Bfr[kc][2 * p + 1] = xr[ko + 1];
    }
  }
  const float* xrb = xin + (size_t)b * COUT * HW;
  float* ob = out + (size_t)b * COUT * HW;
#pragma unroll
  for (int it = 0; it < 4; ++it) {
    int mt = it * 4 + wave;
    v8f acc = {};
#pragma unroll
    for (int kc = 0; kc < 2; ++kc) {
      v16h A = *(const v16h*)(w3f + (((mt * 2 + kc) * 32 + lane) << 4));
      acc = __builtin_amdgcn_wmma_f32_16x16x32_f16(false, A, false, Bfr[kc],
                                                   (short)0, acc, false, false);
    }
#pragma unroll
    for (int i = 0; i < 8; ++i) {
      int m = mt * 16 + (hi << 3) + i;
      float v = acc[i] * g3[m] + b3[m];
      v = v > 0.f ? v : 0.f;                      // relu BEFORE residual add
      ob[m * HW + n0 + n] = v + xrb[m * HW + n0 + n];
    }
  }
}

// ---------------------------------------------------------------------------
// Grouped 3x3 conv (groups=4, 16->16 each) + BN + relu  ->  k
// ---------------------------------------------------------------------------
__global__ void k_keyembed(const float* __restrict__ x1, const float* __restrict__ kew,
                           const float* __restrict__ keg, const float* __restrict__ keb,
                           float* __restrict__ kout) {
  int idx = blockIdx.x;
  int yrow = idx % Hh; idx /= Hh;
  int o = idx % Cc;    int b = idx / Cc;
  int xcol = threadIdx.x;
  if (xcol >= Ww) return;
  int g = o >> 4;
  const float* xb = x1 + ((size_t)b * Cc + g * 16) * HW;
  const float* wo = kew + o * 16 * 9;
  float s = 0.f;
  for (int ci = 0; ci < 16; ++ci) {
    const float* xc = xb + ci * HW;
    const float* wc = wo + ci * 9;
#pragma unroll
    for (int ky = 0; ky < 3; ++ky) {
      int yy = yrow + ky - 1;
      if (yy < 0 || yy >= Hh) continue;
#pragma unroll
      for (int kx = 0; kx < 3; ++kx) {
        int xx = xcol + kx - 1;
        if (xx < 0 || xx >= Ww) continue;
        s += xc[yy * Ww + xx] * wc[ky * 3 + kx];
      }
    }
  }
  s = s * keg[o] + keb[o];
  kout[((size_t)b * Cc + o) * HW + yrow * Ww + xcol] = s > 0.f ? s : 0.f;
}

// ---------------------------------------------------------------------------
// embed stage 1: wmid[b][32][n] = relu(bn(em_w1(32x128) @ concat(x1,k)))
// one thread per (b,n) accumulating all 32 outputs; weights in LDS
// ---------------------------------------------------------------------------
__global__ void __launch_bounds__(256)
k_embed1(const float* __restrict__ x1, const float* __restrict__ kbuf,
         const float* __restrict__ w, const float* __restrict__ ga,
         const float* __restrict__ bb, float* __restrict__ wmid) {
  __shared__ float Ws[CEM * 128];
  for (int i = threadIdx.x; i < CEM * 128; i += 256) Ws[i] = w[i];
  __syncthreads();
  int t = blockIdx.x * 256 + threadIdx.x;
  if (t >= Bb * HW) return;
  int n = t % HW, b = t / HW;
  const float* xb = x1   + (size_t)b * Cc * HW + n;
  const float* kb = kbuf + (size_t)b * Cc * HW + n;
  float s[CEM];
#pragma unroll
  for (int o = 0; o < CEM; ++o) s[o] = 0.f;
  for (int c = 0; c < 64; ++c) {
    float xv = xb[c * HW];
#pragma unroll
    for (int o = 0; o < CEM; ++o) s[o] += xv * Ws[o * 128 + c];
  }
  for (int c = 0; c < 64; ++c) {
    float kv = kb[c * HW];
#pragma unroll
    for (int o = 0; o < CEM; ++o) s[o] += kv * Ws[o * 128 + 64 + c];
  }
  float* outb = wmid + (size_t)b * CEM * HW + n;
#pragma unroll
  for (int o = 0; o < CEM; ++o) {
    float v = s[o] * ga[o] + bb[o];
    outb[o * HW] = v > 0.f ? v : 0.f;
  }
}

// embed stage 2: wk[b][72][n] = em_w2(72x32) @ wmid + bias  (pre-GN)
__global__ void __launch_bounds__(256)
k_embed2(const float* __restrict__ wmid, const float* __restrict__ w,
         const float* __restrict__ bias, float* __restrict__ wk) {
  __shared__ float Ws[CK * CEM];
  for (int i = threadIdx.x; i < CK * CEM; i += 256) Ws[i] = w[i];
  __syncthreads();
  int t = blockIdx.x * 256 + threadIdx.x;
  if (t >= Bb * HW) return;
  int n = t % HW, b = t / HW;
  const float* mb = wmid + (size_t)b * CEM * HW + n;
  float s[CK];
#pragma unroll
  for (int o = 0; o < CK; ++o) s[o] = 0.f;
  for (int c = 0; c < CEM; ++c) {
    float mv = mb[c * HW];
#pragma unroll
    for (int o = 0; o < CK; ++o) s[o] += mv * Ws[o * CEM + c];
  }
  float* ob = wk + (size_t)b * CK * HW + n;
#pragma unroll
  for (int o = 0; o < CK; ++o) ob[o * HW] = s[o] + bias[o];
}

// GroupNorm stats: per (b, g) over 9 channels x HW -> mean, rstd
__global__ void __launch_bounds__(256)
k_gnstats(const float* __restrict__ wk, float* __restrict__ stats) {
  __shared__ float r1[256], r2[256];
  int bg = blockIdx.x;                 // b*8+g
  int b = bg >> 3, g = bg & 7;
  const float* base = wk + ((size_t)b * CK + g * 9) * HW;
  const int NE = 9 * HW;
  float s = 0.f, s2 = 0.f;
  for (int i = threadIdx.x; i < NE; i += 256) { float v = base[i]; s += v; s2 += v * v; }
  r1[threadIdx.x] = s; r2[threadIdx.x] = s2;
  __syncthreads();
  for (int st = 128; st > 0; st >>= 1) {
    if (threadIdx.x < st) { r1[threadIdx.x] += r1[threadIdx.x + st];
                            r2[threadIdx.x] += r2[threadIdx.x + st]; }
    __syncthreads();
  }
  if (threadIdx.x == 0) {
    float mean = r1[0] / (float)NE;
    float var  = r2[0] / (float)NE - mean * mean;
    stats[bg * 2]     = mean;
    stats[bg * 2 + 1] = rsqrtf(var + 1e-5f);
  }
}

// Local dynamic conv (9-tap, kernel shared over 8 channels) + BN + SiLU -> y
__global__ void __launch_bounds__(256)
k_localconv(const float* __restrict__ v, const float* __restrict__ wk,
            const float* __restrict__ stats, const float* __restrict__ gng,
            const float* __restrict__ gnb, const float* __restrict__ bng,
            const float* __restrict__ bnb, float* __restrict__ y) {
  int t = blockIdx.x * 256 + threadIdx.x;
  if (t >= Bb * Cc * HW) return;
  int n = t % HW, c = (t / HW) & 63, b = t / (HW * Cc);
  int yrow = n / Ww, xcol = n % Ww;
  int cg = c >> 3;
  const float* vb = v  + ((size_t)b * Cc + c) * HW;
  const float* wb = wk + ((size_t)b * CK + cg * 9) * HW + n;
  float mean = stats[(b * 8 + cg) * 2];
  float rstd = stats[(b * 8 + cg) * 2 + 1];
  float s = 0.f;
#pragma unroll
  for (int t9 = 0; t9 < 9; ++t9) {
    int yy = yrow + t9 / 3 - 1, xx = xcol + t9 % 3 - 1;
    float pv = (yy >= 0 && yy < Hh && xx >= 0 && xx < Ww) ? vb[yy * Ww + xx] : 0.f;
    int ch = cg * 9 + t9;
    float wv = (wb[t9 * HW] - mean) * rstd * gng[ch] + gnb[ch];
    s += pv * wv;
  }
  s = s * bng[c] + bnb[c];
  y[t] = s / (1.f + __expf(-s));                  // SiLU
}

// GAP over HW of (y + k) -> gap[b][c]
__global__ void __launch_bounds__(256)
k_gap(const float* __restrict__ y, const float* __restrict__ kbuf,
      float* __restrict__ gap) {
  __shared__ float red[256];
  int bc = blockIdx.x;
  const float* yb = y    + (size_t)bc * HW;
  const float* kb = kbuf + (size_t)bc * HW;
  float s = 0.f;
  for (int i = threadIdx.x; i < HW; i += 256) s += yb[i] + kb[i];
  red[threadIdx.x] = s;
  __syncthreads();
  for (int st = 128; st > 0; st >>= 1) {
    if (threadIdx.x < st) red[threadIdx.x] += red[threadIdx.x + st];
    __syncthreads();
  }
  if (threadIdx.x == 0) gap[bc] = red[0] / (float)HW;
}

// SE attention: gap -> 32 -> 128 -> pairwise softmax -> attn[b][64][2]
__global__ void __launch_bounds__(128)
k_attn(const float* __restrict__ gap, const float* __restrict__ w1,
       const float* __restrict__ b1, const float* __restrict__ gg,
       const float* __restrict__ gb, const float* __restrict__ w2,
       const float* __restrict__ b2, float* __restrict__ attn) {
  __shared__ float gs[64], a1[32], a2[128];
  int b = blockIdx.x, tid = threadIdx.x;
  if (tid < 64) gs[tid] = gap[b * 64 + tid];
  __syncthreads();
  if (tid < 32) {
    float s = b1[tid];
    for (int c = 0; c < 64; ++c) s += gs[c] * w1[tid * 64 + c];
    s = s * gg[tid] + gb[tid];
    a1[tid] = s > 0.f ? s : 0.f;
  }
  __syncthreads();
  {
    float s = b2[tid];
    for (int j = 0; j < 32; ++j) s += a1[j] * w2[tid * 32 + j];
    a2[tid] = s;
  }
  __syncthreads();
  if (tid < 64) {
    float x0 = a2[2 * tid], x1v = a2[2 * tid + 1];
    float m = fmaxf(x0, x1v);
    float e0 = __expf(x0 - m), e1 = __expf(x1v - m);
    float inv = 1.f / (e0 + e1);
    attn[b * 128 + 2 * tid]     = e0 * inv;
    attn[b * 128 + 2 * tid + 1] = e1 * inv;
  }
}

// radix combine: cot = y*attn0 + k*attn1
__global__ void __launch_bounds__(256)
k_cot(const float* __restrict__ y, const float* __restrict__ kbuf,
      const float* __restrict__ attn, float* __restrict__ cot) {
  int t = blockIdx.x * 256 + threadIdx.x;
  if (t >= Bb * Cc * HW) return;
  int c = (t / HW) & 63, b = t / (HW * Cc);
  float a0 = attn[b * 128 + 2 * c], a1v = attn[b * 128 + 2 * c + 1];
  cot[t] = y[t] * a0 + kbuf[t] * a1v;
}

// ---------------------------------------------------------------------------
extern "C" void kernel_launch(void* const* d_in, const int* in_sizes, int n_in,
                              void* d_out, int out_size, void* d_ws, size_t ws_size,
                              hipStream_t stream) {
  const float* x     = (const float*)d_in[0];
  const float* w1    = (const float*)d_in[1];
  const float* g1    = (const float*)d_in[2];
  const float* b1    = (const float*)d_in[3];
  const float* kew   = (const float*)d_in[4];
  const float* keg   = (const float*)d_in[5];
  const float* keb   = (const float*)d_in[6];
  const float* emw1  = (const float*)d_in[7];
  const float* emg1  = (const float*)d_in[8];
  const float* emb1  = (const float*)d_in[9];
  const float* emw2  = (const float*)d_in[10];
  const float* embi2 = (const float*)d_in[11];
  const float* gng   = (const float*)d_in[12];
  const float* gnb   = (const float*)d_in[13];
  const float* c1w   = (const float*)d_in[14];
  const float* c1g   = (const float*)d_in[15];
  const float* c1b   = (const float*)d_in[16];
  const float* bng   = (const float*)d_in[17];
  const float* bnb   = (const float*)d_in[18];
  const float* sew1  = (const float*)d_in[19];
  const float* seb1  = (const float*)d_in[20];
  const float* seg   = (const float*)d_in[21];
  const float* sebb  = (const float*)d_in[22];
  const float* sew2  = (const float*)d_in[23];
  const float* seb2  = (const float*)d_in[24];
  const float* w3    = (const float*)d_in[25];
  const float* g3    = (const float*)d_in[26];
  const float* b3    = (const float*)d_in[27];
  float* out = (float*)d_out;

  char* ws = (char*)d_ws;
  const size_t S1 = (size_t)Bb * Cc * HW * 4;     // 25,690,112 B
  float*    x1    = (float*)(ws);
  float*    kbuf  = (float*)(ws + S1);
  float*    wmid  = (float*)(ws + 2 * S1);
  float*    wk    = (float*)(ws + 2 * S1 + S1 / 2);
  const size_t Swk = (size_t)Bb * CK * HW * 4;    // 28,901,376 B
  float*    vbuf  = (float*)(ws + 2 * S1 + S1 / 2 + Swk);
  float*    ybuf  = (float*)((char*)vbuf + S1);
  float*    stats = (float*)((char*)ybuf + S1);
  float*    gapb  = (float*)((char*)stats + 2048);
  float*    attnb = (float*)((char*)gapb + 8192);
  _Float16* w1f   = (_Float16*)((char*)attnb + 16384);
  _Float16* c1f   = (_Float16*)((char*)w1f + 32768);
  _Float16* w3f   = (_Float16*)((char*)c1f + 8192);

  // 1. weight fragment swizzle
  prep_frags<<<(36864 + 255) / 256, 256, 0, stream>>>(w1, c1w, w3, w1f, c1f, w3f);
  // 2. conv1x1 reduce 256->64 + BN (WMMA)
  gemm64<256><<<Bb * NTt, 128, 0, stream>>>(x, w1f, g1, b1, x1);
  // 3. grouped 3x3 key-embed + BN + relu
  k_keyembed<<<Bb * Cc * Hh, 64, 0, stream>>>(x1, kew, keg, keb, kbuf);
  // 4. embed MLP stage 1 (128->32) + BN + relu
  k_embed1<<<(Bb * HW) / 256, 256, 0, stream>>>(x1, kbuf, emw1, emg1, emb1, wmid);
  // 5. embed MLP stage 2 (32->72) + bias
  k_embed2<<<(Bb * HW) / 256, 256, 0, stream>>>(wmid, emw2, embi2, wk);
  // 6. GroupNorm stats
  k_gnstats<<<Bb * 8, 256, 0, stream>>>(wk, stats);
  // 7. value conv 64->64 + BN (WMMA)
  gemm64<64><<<Bb * NTt, 128, 0, stream>>>(x1, c1f, c1g, c1b, vbuf);
  // 8. local dynamic conv + GN apply + BN + SiLU
  k_localconv<<<(Bb * Cc * HW) / 256, 256, 0, stream>>>(vbuf, wk, stats, gng, gnb,
                                                        bng, bnb, ybuf);
  // 9. global average pool of (y + k)
  k_gap<<<Bb * Cc, 256, 0, stream>>>(ybuf, kbuf, gapb);
  // 10. SE attention + pairwise softmax
  k_attn<<<Bb, 128, 0, stream>>>(gapb, sew1, seb1, seg, sebb, sew2, seb2, attnb);
  // 11. radix combine (writes into vbuf, which is free now)
  k_cot<<<(Bb * Cc * HW) / 256, 256, 0, stream>>>(ybuf, kbuf, attnb, vbuf);
  // 12. conv1x1 expand 64->256 + BN + relu + residual (WMMA)
  k_expand<<<Bb * NTt, 128, 0, stream>>>(vbuf, w3f, g3, b3, x, out);
  (void)in_sizes; (void)n_in; (void)out_size; (void)ws_size;
}
